// PathRNNEncoder_46239617909503
// MI455X (gfx1250) — compile-verified, matrix operand order checked by reading, not run on hardware
//
#include <hip/hip_runtime.h>

// ---------------------------------------------------------------------------
// Problem constants (match reference)
// ---------------------------------------------------------------------------
constexpr int T_TOK   = 16;
constexpr int N_UNITS = 4096;
constexpr int E       = 128;
constexpr int Hh      = 128;   // hidden
constexpr int FOURH   = 512;
constexpr int L_PATH  = 32;
constexpr int N_PATHS = 1024;
constexpr int Bsz     = 8;

// ---------------------------------------------------------------------------
// Vector types for WMMA
// ---------------------------------------------------------------------------
typedef __attribute__((ext_vector_type(16))) __bf16 v16bf;
typedef __attribute__((ext_vector_type(8)))  __bf16 v8bf;
typedef __attribute__((ext_vector_type(8)))  float  v8f;

// Load one lane's slice of a 16x32 bf16 tile (row-major source).
// Lane layout (CDNA5 16-bit A/B 16x32): lanes 0-15 hold K={kb..kb+7, kb+16..kb+23},
// lanes 16-31 hold K={kb+8..kb+15, kb+24..kb+31}.
__device__ __forceinline__ v16bf load_k32(const __bf16* __restrict__ rowptr, int kbase, int hi) {
    const v8bf lo  = *(const v8bf*)(rowptr + kbase + hi * 8);
    const v8bf hi8 = *(const v8bf*)(rowptr + kbase + 16 + hi * 8);
    v16bf r;
#pragma unroll
    for (int j = 0; j < 8; ++j) { r[j] = lo[j]; r[8 + j] = hi8[j]; }
    return r;
}

// ---------------------------------------------------------------------------
// WMMA GEMM:  C[M x N] = A1(M x K1) @ W1^T + [A2(M x K2) @ W2^T] + bias
//   - A1/A2/W1/W2 bf16 row-major, W row = output feature (so @ W^T).
//   - If lens != nullptr: A1 row n reads time slice t = clip(lens[n]-1-step, 0, Tmax-1)
//     from a [T][M][K1] tensor at base A1 (backward-LSTM reversed gather).
//   - Block = 64 threads (2 waves); each wave computes a 32x64 tile
//     (2 M-subtiles x 4 N-subtiles = 8 accumulators; each B fragment feeds
//     two WMMAs, each A fragment feeds four -> 1.5 loads/WMMA).
//   - Grid = (M/32, N/128).
// ---------------------------------------------------------------------------
__global__ void wmma_gemm_kernel(const __bf16* __restrict__ A1, int K1, const __bf16* __restrict__ W1,
                                 const __bf16* __restrict__ A2, int K2, const __bf16* __restrict__ W2,
                                 const float* __restrict__ bias, float* __restrict__ C,
                                 int M, int N,
                                 const int* __restrict__ lens, int step, int Tmax)
{
    const int lane = threadIdx.x & 31;
    const int wave = threadIdx.x >> 5;
    const int r    = lane & 15;
    const int hi   = lane >> 4;
    const int m0   = blockIdx.x * 32;
    const int n0   = blockIdx.y * 128 + wave * 64;

    v8f acc[2][4];
#pragma unroll
    for (int u = 0; u < 2; ++u)
#pragma unroll
        for (int s = 0; s < 4; ++s)
#pragma unroll
            for (int j = 0; j < 8; ++j) acc[u][s][j] = 0.f;

    // Per-lane A row pointers for the two 16-row M-subtiles.
    const __bf16* a1p[2];
#pragma unroll
    for (int u = 0; u < 2; ++u) {
        const int arow = m0 + u * 16 + r;
        if (lens) {
            int t = lens[arow] - 1 - step;
            t = t < 0 ? 0 : (t > Tmax - 1 ? Tmax - 1 : t);
            a1p[u] = A1 + ((size_t)t * M + arow) * K1;
        } else {
            a1p[u] = A1 + (size_t)arow * K1;
        }
    }

    for (int kt = 0; kt < K1; kt += 32) {
        v16bf a0 = load_k32(a1p[0], kt, hi);
        v16bf a1 = load_k32(a1p[1], kt, hi);
#pragma unroll
        for (int s = 0; s < 4; ++s) {
            const __bf16* wp = W1 + (size_t)(n0 + s * 16 + r) * K1;
            v16bf b = load_k32(wp, kt, hi);
            acc[0][s] = __builtin_amdgcn_wmma_f32_16x16x32_bf16(false, a0, false, b,
                                                                (short)0, acc[0][s], false, false);
            acc[1][s] = __builtin_amdgcn_wmma_f32_16x16x32_bf16(false, a1, false, b,
                                                                (short)0, acc[1][s], false, false);
        }
    }
    if (A2) {
        const __bf16* a2p0 = A2 + (size_t)(m0 + r) * K2;
        const __bf16* a2p1 = A2 + (size_t)(m0 + 16 + r) * K2;
        for (int kt = 0; kt < K2; kt += 32) {
            v16bf a0 = load_k32(a2p0, kt, hi);
            v16bf a1 = load_k32(a2p1, kt, hi);
#pragma unroll
            for (int s = 0; s < 4; ++s) {
                const __bf16* wp = W2 + (size_t)(n0 + s * 16 + r) * K2;
                v16bf b = load_k32(wp, kt, hi);
                acc[0][s] = __builtin_amdgcn_wmma_f32_16x16x32_bf16(false, a0, false, b,
                                                                    (short)0, acc[0][s], false, false);
                acc[1][s] = __builtin_amdgcn_wmma_f32_16x16x32_bf16(false, a1, false, b,
                                                                    (short)0, acc[1][s], false, false);
            }
        }
    }
    // D layout: VGPR v -> row m0 + u*16 + hi*8 + v, col n0 + s*16 + r
#pragma unroll
    for (int s = 0; s < 4; ++s) {
        const int n  = n0 + s * 16 + r;
        const float bv = bias ? bias[n] : 0.f;
#pragma unroll
        for (int u = 0; u < 2; ++u) {
#pragma unroll
            for (int v = 0; v < 8; ++v) {
                const int m = m0 + u * 16 + hi * 8 + v;
                C[(size_t)m * N + n] = acc[u][s][v] + bv;
            }
        }
    }
}

// ---------------------------------------------------------------------------
// Elementwise LSTM cell: gates -> c,h ; writes h (bf16) into recurrent state
// and masked into the concatenated sequence buffer hseq[T][M][256].
// dir==0 (forward):  hseq[step][n][coloff+h] = (step<len ? h : 0)
// dir==1 (backward): if step<len: hseq[len-1-step][n][coloff+h] = h
// ---------------------------------------------------------------------------
__device__ __forceinline__ float sigf(float x) { return 1.f / (1.f + expf(-x)); }

__global__ void lstm_cell_kernel(const float* __restrict__ G, const float* __restrict__ bias,
                                 float* __restrict__ c_state, __bf16* __restrict__ h_bf,
                                 __bf16* __restrict__ hseq, const int* __restrict__ lens,
                                 int step, int dir, int coloff, int M)
{
    const int idx = blockIdx.x * blockDim.x + threadIdx.x;
    if (idx >= M * Hh) return;
    const int n = idx >> 7;
    const int h = idx & 127;
    const float* g4 = G + (size_t)n * FOURH;
    const float gi = g4[h]           + bias[h];
    const float gf = g4[Hh + h]      + bias[Hh + h];
    const float gg = g4[2 * Hh + h]  + bias[2 * Hh + h];
    const float go = g4[3 * Hh + h]  + bias[3 * Hh + h];
    float c = c_state[idx];
    c = sigf(gf) * c + sigf(gi) * tanhf(gg);
    c_state[idx] = c;
    const float hv = sigf(go) * tanhf(c);
    h_bf[idx] = (__bf16)hv;
    const int len = lens[n];
    if (dir == 0) {
        const float o = (step < len) ? hv : 0.f;
        hseq[((size_t)step * M + n) * 256 + coloff + h] = (__bf16)o;
    } else if (step < len) {
        const int t = len - 1 - step;
        hseq[((size_t)t * M + n) * 256 + coloff + h] = (__bf16)hv;
    }
}

// ---------------------------------------------------------------------------
// Small utility kernels
// ---------------------------------------------------------------------------
__global__ void f32_to_bf16_kernel(const float* __restrict__ src, __bf16* __restrict__ dst, int n) {
    int i = blockIdx.x * blockDim.x + threadIdx.x;
    if (i < n) dst[i] = (__bf16)src[i];
}

__global__ void tok_len_kernel(const int* __restrict__ units, int* __restrict__ lens) {
    int n = blockIdx.x * blockDim.x + threadIdx.x;
    if (n >= N_UNITS) return;
    int len = T_TOK;
    for (int t = 0; t < T_TOK; ++t) {
        if (units[t * N_UNITS + n] == 0) { len = t; break; }
    }
    lens[n] = len;
}

__global__ void embed_kernel(const int* __restrict__ units, const float* __restrict__ emb,
                             __bf16* __restrict__ x_bf) {
    int idx = blockIdx.x * blockDim.x + threadIdx.x;          // T*NU*E
    if (idx >= T_TOK * N_UNITS * E) return;
    const int e    = idx & 127;
    const int rest = idx >> 7;
    const int n    = rest & (N_UNITS - 1);
    const int t    = rest / N_UNITS;
    const int u    = units[t * N_UNITS + n];
    x_bf[idx] = (__bf16)emb[(size_t)u * E + e];
}

__device__ __forceinline__ float wave_sum(float v) {
#pragma unroll
    for (int off = 16; off > 0; off >>= 1) v += __shfl_xor(v, off, 32);
    return v;
}

// One block (128 threads) per (t,n) row: layernorm -> tanh -> attention score.
__global__ void ln_score_kernel(const float* __restrict__ out, const float* __restrict__ ln_g,
                                const float* __restrict__ ln_b, const float* __restrict__ attn_w,
                                const float* __restrict__ attn_b, const int* __restrict__ lens,
                                float* __restrict__ scores)
{
    const int row = blockIdx.x;                 // t*NU + n
    const int e   = threadIdx.x;                // 0..127
    const int t   = row / N_UNITS;
    const int n   = row - t * N_UNITS;
    const float x = out[(size_t)row * E + e];
    __shared__ float red[4];
    const int wid = e >> 5;

    float s = wave_sum(x);
    if ((e & 31) == 0) red[wid] = s;
    __syncthreads();
    const float mean = (red[0] + red[1] + red[2] + red[3]) * (1.f / 128.f);
    __syncthreads();

    const float d = x - mean;
    s = wave_sum(d * d);
    if ((e & 31) == 0) red[wid] = s;
    __syncthreads();
    const float var = (red[0] + red[1] + red[2] + red[3]) * (1.f / 128.f);
    __syncthreads();

    const float q = tanhf((x - mean) * rsqrtf(var + 1e-5f) * ln_g[e] + ln_b[e]);
    s = wave_sum(q * attn_w[e]);
    if ((e & 31) == 0) red[wid] = s;
    __syncthreads();
    if (e == 0) {
        float sc = red[0] + red[1] + red[2] + red[3] + attn_b[0];
        if (t >= lens[n]) sc = -1e9f;
        scores[(size_t)t * N_UNITS + n] = sc;
    }
}

// One block (128 threads = e) per unit n: softmax over T and weighted pool.
__global__ void softmax_feat_kernel(const float* __restrict__ scores, const float* __restrict__ out,
                                    __bf16* __restrict__ tok_feat)
{
    const int n = blockIdx.x;
    const int e = threadIdx.x;
    float m = -1e30f;
    for (int t = 0; t < T_TOK; ++t) m = fmaxf(m, scores[t * N_UNITS + n]);
    float den = 0.f, acc = 0.f;
    for (int t = 0; t < T_TOK; ++t) {
        const float w = expf(scores[t * N_UNITS + n] - m);
        den += w;
        acc += w * out[((size_t)t * N_UNITS + n) * E + e];
    }
    tok_feat[(size_t)n * E + e] = (__bf16)(acc / den);
}

// One block (128 threads = e) per path p: gather path features + path length.
__global__ void path_prep_kernel(const int* __restrict__ paths, const int* __restrict__ upd,
                                 const int* __restrict__ ppd, const __bf16* __restrict__ tok_feat,
                                 __bf16* __restrict__ feats, int* __restrict__ p_len)
{
    const int p = blockIdx.x;
    const int e = threadIdx.x;
    __shared__ int s_pl[L_PATH];
    __shared__ int s_gidx[L_PATH];
    __shared__ int s_keep[L_PATH];
    if (e < L_PATH) s_pl[e] = paths[e * N_PATHS + p];
    __syncthreads();
    if (e == 0) {
        int d = 0, accum = 0;
        for (int j = 0; j < Bsz; ++j) { accum += ppd[j]; if (p < accum) { d = j; break; } }
        int off = 0;
        for (int j = 0; j < d; ++j) off += upd[j];
        const int un = upd[d];
        int fe = L_PATH;
        for (int l = 0; l < L_PATH; ++l) { if (s_pl[l] == -1) { fe = l; break; } }
        int plen = L_PATH; bool found = false;
        for (int l = 0; l < L_PATH; ++l) {
            const int pl = s_pl[l];
            const bool within = l < fe;
            const bool keep = within && (pl >= 0) && (pl < un);
            int g = (pl < 0 ? 0 : pl) + off;
            if (g > N_UNITS - 1) g = N_UNITS - 1;
            if (g < 0) g = 0;
            s_gidx[l] = g;
            s_keep[l] = keep ? 1 : 0;
            const bool mask = within ? ((pl < 0) || (pl > un)) : true;
            if (mask && !found) { plen = l; found = true; }
        }
        p_len[p] = plen;
    }
    __syncthreads();
    for (int l = 0; l < L_PATH; ++l) {
        __bf16 v = s_keep[l] ? tok_feat[(size_t)s_gidx[l] * E + e] : (__bf16)0.f;
        feats[((size_t)l * N_PATHS + p) * E + e] = v;
    }
}

// ---------------------------------------------------------------------------
// Host driver
// ---------------------------------------------------------------------------
static inline void launch_gemm(const __bf16* A1, int K1, const __bf16* W1,
                               const __bf16* A2, int K2, const __bf16* W2,
                               const float* bias, float* C, int M, int N,
                               const int* lens, int step, int Tmax, hipStream_t st)
{
    dim3 g(M / 32, N / 128);
    wmma_gemm_kernel<<<g, 64, 0, st>>>(A1, K1, W1, A2, K2, W2, bias, C, M, N, lens, step, Tmax);
}

extern "C" void kernel_launch(void* const* d_in, const int* in_sizes, int n_in,
                              void* d_out, int out_size, void* d_ws, size_t ws_size,
                              hipStream_t stream)
{
    // ---- inputs ----
    const int*   units   = (const int*)  d_in[0];
    const int*   paths   = (const int*)  d_in[1];
    const int*   upd     = (const int*)  d_in[2];
    const int*   ppd     = (const int*)  d_in[3];
    const float* emb     = (const float*)d_in[4];
    const float* tl_Wif  = (const float*)d_in[5];
    const float* tl_Whf  = (const float*)d_in[6];
    const float* tl_bf   = (const float*)d_in[7];
    const float* tl_Wib  = (const float*)d_in[8];
    const float* tl_Whb  = (const float*)d_in[9];
    const float* tl_bb   = (const float*)d_in[10];
    const float* lin_W   = (const float*)d_in[11];
    const float* lin_b   = (const float*)d_in[12];
    const float* ln_g    = (const float*)d_in[13];
    const float* ln_b    = (const float*)d_in[14];
    const float* attn_w  = (const float*)d_in[15];
    const float* attn_b  = (const float*)d_in[16];
    const float* pl_Wif  = (const float*)d_in[17];
    const float* pl_Whf  = (const float*)d_in[18];
    const float* pl_bf   = (const float*)d_in[19];
    const float* pl_Wib  = (const float*)d_in[20];
    const float* pl_Whb  = (const float*)d_in[21];
    const float* pl_bb   = (const float*)d_in[22];
    const float* ul_W    = (const float*)d_in[23];
    const float* ul_b    = (const float*)d_in[24];
    float* outp = (float*)d_out;

    // ---- workspace carve-out ----
    char* base = (char*)d_ws;
    size_t off = 0;
    auto alloc = [&](size_t bytes) -> char* {
        char* p = base + off;
        off = (off + bytes + 255) & ~(size_t)255;
        return p;
    };
    __bf16* x_bf      = (__bf16*)alloc((size_t)T_TOK * N_UNITS * E * 2);
    int*    tok_len   = (int*)   alloc((size_t)N_UNITS * 4);
    __bf16* wb_tlWif  = (__bf16*)alloc(FOURH * E * 2);
    __bf16* wb_tlWhf  = (__bf16*)alloc(FOURH * Hh * 2);
    __bf16* wb_tlWib  = (__bf16*)alloc(FOURH * E * 2);
    __bf16* wb_tlWhb  = (__bf16*)alloc(FOURH * Hh * 2);
    __bf16* wb_linW   = (__bf16*)alloc(E * 2 * Hh * 2);
    __bf16* wb_plWif  = (__bf16*)alloc(FOURH * E * 2);
    __bf16* wb_plWhf  = (__bf16*)alloc(FOURH * Hh * 2);
    __bf16* wb_plWib  = (__bf16*)alloc(FOURH * E * 2);
    __bf16* wb_plWhb  = (__bf16*)alloc(FOURH * Hh * 2);
    __bf16* wb_ulW    = (__bf16*)alloc(E * 2 * Hh * 2);
    __bf16* hcat_tok  = (__bf16*)alloc((size_t)T_TOK * N_UNITS * 256 * 2);
    float*  Gbuf      = (float*) alloc((size_t)N_UNITS * FOURH * 4);
    float*  c_state   = (float*) alloc((size_t)N_UNITS * Hh * 4);
    __bf16* h_bf      = (__bf16*)alloc((size_t)N_UNITS * Hh * 2);
    float*  out_lin   = (float*) alloc((size_t)T_TOK * N_UNITS * E * 4);
    float*  scores    = (float*) alloc((size_t)T_TOK * N_UNITS * 4);
    __bf16* tok_feat  = (__bf16*)alloc((size_t)N_UNITS * E * 2);
    __bf16* feats     = (__bf16*)alloc((size_t)L_PATH * N_PATHS * E * 2);
    int*    p_len     = (int*)   alloc((size_t)N_PATHS * 4);
    __bf16* hcat_path = (__bf16*)alloc((size_t)L_PATH * N_PATHS * 256 * 2);
    (void)ws_size; (void)in_sizes; (void)n_in; (void)out_size;

    auto conv = [&](const float* s, __bf16* d, int n) {
        f32_to_bf16_kernel<<<(n + 255) / 256, 256, 0, stream>>>(s, d, n);
    };

    // ---- weight conversion ----
    conv(tl_Wif, wb_tlWif, FOURH * E);
    conv(tl_Whf, wb_tlWhf, FOURH * Hh);
    conv(tl_Wib, wb_tlWib, FOURH * E);
    conv(tl_Whb, wb_tlWhb, FOURH * Hh);
    conv(lin_W,  wb_linW,  E * 2 * Hh);
    conv(pl_Wif, wb_plWif, FOURH * E);
    conv(pl_Whf, wb_plWhf, FOURH * Hh);
    conv(pl_Wib, wb_plWib, FOURH * E);
    conv(pl_Whb, wb_plWhb, FOURH * Hh);

    // ---- token lengths + embedding ----
    tok_len_kernel<<<(N_UNITS + 255) / 256, 256, 0, stream>>>(units, tok_len);
    {
        int n = T_TOK * N_UNITS * E;
        embed_kernel<<<(n + 255) / 256, 256, 0, stream>>>(units, emb, x_bf);
    }

    hipMemsetAsync(hcat_tok, 0, (size_t)T_TOK * N_UNITS * 256 * 2, stream);

    const int cellBlocks = (N_UNITS * Hh + 255) / 256;
    // ---- token forward LSTM ----
    hipMemsetAsync(c_state, 0, (size_t)N_UNITS * Hh * 4, stream);
    hipMemsetAsync(h_bf,    0, (size_t)N_UNITS * Hh * 2, stream);
    for (int t = 0; t < T_TOK; ++t) {
        launch_gemm(x_bf + (size_t)t * N_UNITS * E, E, wb_tlWif,
                    h_bf, Hh, wb_tlWhf, nullptr, Gbuf, N_UNITS, FOURH,
                    nullptr, 0, 0, stream);
        lstm_cell_kernel<<<cellBlocks, 256, 0, stream>>>(Gbuf, tl_bf, c_state, h_bf,
                                                         hcat_tok, tok_len, t, 0, 0, N_UNITS);
    }
    // ---- token backward LSTM (reversed per-row gather) ----
    hipMemsetAsync(c_state, 0, (size_t)N_UNITS * Hh * 4, stream);
    hipMemsetAsync(h_bf,    0, (size_t)N_UNITS * Hh * 2, stream);
    for (int s = 0; s < T_TOK; ++s) {
        launch_gemm(x_bf, E, wb_tlWib,
                    h_bf, Hh, wb_tlWhb, nullptr, Gbuf, N_UNITS, FOURH,
                    tok_len, s, T_TOK, stream);
        lstm_cell_kernel<<<cellBlocks, 256, 0, stream>>>(Gbuf, tl_bb, c_state, h_bf,
                                                         hcat_tok, tok_len, s, 1, Hh, N_UNITS);
    }

    // ---- lin projection: out = hcat @ lin_W^T + lin_b ----
    launch_gemm(hcat_tok, 256, wb_linW, nullptr, 0, nullptr,
                lin_b, out_lin, T_TOK * N_UNITS, E, nullptr, 0, 0, stream);

    // ---- layernorm + tanh + attention scores, softmax pooling ----
    ln_score_kernel<<<T_TOK * N_UNITS, 128, 0, stream>>>(out_lin, ln_g, ln_b, attn_w, attn_b,
                                                         tok_len, scores);
    softmax_feat_kernel<<<N_UNITS, 128, 0, stream>>>(scores, out_lin, tok_feat);

    // ---- path feature gather ----
    path_prep_kernel<<<N_PATHS, 128, 0, stream>>>(paths, upd, ppd, tok_feat, feats, p_len);

    hipMemsetAsync(hcat_path, 0, (size_t)L_PATH * N_PATHS * 256 * 2, stream);

    const int cellBlocksP = (N_PATHS * Hh + 255) / 256;
    // ---- path forward LSTM ----
    hipMemsetAsync(c_state, 0, (size_t)N_PATHS * Hh * 4, stream);
    hipMemsetAsync(h_bf,    0, (size_t)N_PATHS * Hh * 2, stream);
    for (int t = 0; t < L_PATH; ++t) {
        launch_gemm(feats + (size_t)t * N_PATHS * E, E, wb_plWif,
                    h_bf, Hh, wb_plWhf, nullptr, Gbuf, N_PATHS, FOURH,
                    nullptr, 0, 0, stream);
        lstm_cell_kernel<<<cellBlocksP, 256, 0, stream>>>(Gbuf, pl_bf, c_state, h_bf,
                                                          hcat_path, p_len, t, 0, 0, N_PATHS);
    }
    // ---- path backward LSTM ----
    hipMemsetAsync(c_state, 0, (size_t)N_PATHS * Hh * 4, stream);
    hipMemsetAsync(h_bf,    0, (size_t)N_PATHS * Hh * 2, stream);
    for (int s = 0; s < L_PATH; ++s) {
        launch_gemm(feats, E, wb_plWib,
                    h_bf, Hh, wb_plWhb, nullptr, Gbuf, N_PATHS, FOURH,
                    p_len, s, L_PATH, stream);
        lstm_cell_kernel<<<cellBlocksP, 256, 0, stream>>>(Gbuf, pl_bb, c_state, h_bf,
                                                          hcat_path, p_len, s, 1, Hh, N_PATHS);
    }

    // ---- final projection: d_out = hcat_path @ ul_W^T + ul_b ----
    conv(ul_W, wb_ulW, E * 2 * Hh);
    launch_gemm(hcat_path, 256, wb_ulW, nullptr, 0, nullptr,
                ul_b, outp, L_PATH * N_PATHS, E, nullptr, 0, 0, stream);
}
